// MatlabBlockNCB_Linear_56264071577751
// MI455X (gfx1250) — matrix operand order repaired and checked by reading, not compiled, not run on hardware
//
#include <hip/hip_runtime.h>

// ---------------------------------------------------------------------------
// CDNA5 (gfx1250) bf16 WMMA pipeline for the cosine-attention reference.
//   K = Wk X + bk ; Q = Wq X + bq        (WMMA bf16 GEMM, fp32 acc)
//   Y = (Qt K) * rsqrt(max(|Q|^2 |K|^2, eps))
//   SM = softmax over query axis (column softmax of [b,k])
//   Z = X SM                             (WMMA bf16 GEMM, fp32 acc -> d_out)
// A-tile staging uses GLOBAL_LOAD_ASYNC_TO_LDS_B128 (ASYNCcnt-tracked).
// ---------------------------------------------------------------------------

typedef __attribute__((ext_vector_type(16))) __bf16 v16bf;
typedef __attribute__((ext_vector_type(8)))  float  v8f;

#define EPS_F 1e-12f

__device__ __forceinline__ unsigned short f2bf_bits(float f) {
  unsigned int u = __float_as_uint(f);
  unsigned int lsb = (u >> 16) & 1u;
  u += 0x7fffu + lsb;                      // round-to-nearest-even
  return (unsigned short)(u >> 16);
}
__device__ __forceinline__ float bf2f(unsigned short u) {
  return __uint_as_float(((unsigned int)u) << 16);
}

// Two aligned 16B LDS loads -> one 16x bf16 WMMA fragment, no b16 repacking.
union FragCvt {
  uint4 q[2];
  v16bf v;
};

// ---------------------------------------------------------------------------
// fp32 -> bf16 convert (grid-stride)
// ---------------------------------------------------------------------------
__global__ void k_f32_to_bf16(const float* __restrict__ in,
                              unsigned short* __restrict__ out, int nElem) {
  for (int i = blockIdx.x * blockDim.x + threadIdx.x; i < nElem;
       i += gridDim.x * blockDim.x)
    out[i] = f2bf_bits(in[i]);
}

// ---------------------------------------------------------------------------
// Generic 128x128-tile bf16 WMMA GEMM, fp32 accumulate.
//   A: [M,K] row-major bf16 (ushort bits), lda = row stride
//   B: [K,N] row-major bf16, ldb = row stride (staged transposed in LDS)
// mode 0: v += bias[m]; bf16 store to outBf [M,N] and/or outBfT [N,M]
// mode 1: v *= rsqrt(max(dq2[m]*dk2[n], eps)); f32 store to outF
// mode 2: f32 store to outF
// 256 threads = 8 waves (4 along M x 2 along N); wave tile 32x64 = 2x4 WMMA.
// ---------------------------------------------------------------------------
__global__ __launch_bounds__(256)
void k_wmma_gemm(const unsigned short* __restrict__ A, int lda,
                 const unsigned short* __restrict__ Bm, int ldb,
                 int M, int N, int K, int mode,
                 const float* __restrict__ bias,
                 unsigned short* __restrict__ outBf,
                 unsigned short* __restrict__ outBfT,
                 float* __restrict__ outF,
                 const float* __restrict__ dq2,
                 const float* __restrict__ dk2) {
  // Row stride 40 ushorts = 80 B: 16B-aligned rows, conflict-spread banks.
  __shared__ unsigned short As[128][40];    // [m][k]  128x32 tile
  __shared__ unsigned short BsT[128][40];   // [n][k]  32x128 tile, transposed

  const int tid   = threadIdx.x;
  const int lane  = tid & 31;
  const int wave  = tid >> 5;
  const int waveM = wave & 3;               // 0..3 -> M offset *32
  const int waveN = wave >> 2;              // 0..1 -> N offset *64
  const int bm = blockIdx.y * 128;
  const int bn = blockIdx.x * 128;

  // B-staging decomposition: thread owns K-pair kp (k=2kp,2kp+1), 8 N columns.
  const int kp = tid >> 4;                  // 0..15
  const int no = (tid & 15) * 8;            // 0..120

  v8f acc[2][4];
  for (int i = 0; i < 2; ++i)
    for (int j = 0; j < 4; ++j)
      for (int e = 0; e < 8; ++e) acc[i][j][e] = 0.f;

  for (int kt = 0; kt < K; kt += 32) {
    // ---- Stage A tile 128x32 via async global->LDS copies (ASYNCcnt),
    //      16B per lane, no VGPR round trip (512 chunks / 256 threads).
#pragma unroll
    for (int it = 0; it < 2; ++it) {
      int chunk = tid + it * 256;
      int r = chunk >> 2;
      int c = (chunk & 3) * 8;
      unsigned ldsOff = (unsigned)(size_t)(&As[r][c]);   // LDS byte offset
      const unsigned short* gp = &A[(size_t)(bm + r) * lda + kt + c];
      asm volatile("global_load_async_to_lds_b128 %0, %1, off"
                   :: "v"(ldsOff), "v"(gp)
                   : "memory");
    }
    // ---- Stage B tile 32x128, transposed to [n][k]: two row-chunks in,
    //      8 packed {k,k+1} u32 stores out (contiguous in BsT rows).
    {
      uint4 r0 = *reinterpret_cast<const uint4*>(
          &Bm[(size_t)(kt + 2 * kp) * ldb + bn + no]);
      uint4 r1 = *reinterpret_cast<const uint4*>(
          &Bm[(size_t)(kt + 2 * kp + 1) * ldb + bn + no]);
      const unsigned short* p0 = reinterpret_cast<const unsigned short*>(&r0);
      const unsigned short* p1 = reinterpret_cast<const unsigned short*>(&r1);
#pragma unroll
      for (int i = 0; i < 8; ++i) {
        unsigned int pk = (unsigned int)p0[i] | ((unsigned int)p1[i] << 16);
        *reinterpret_cast<unsigned int*>(&BsT[no + i][2 * kp]) = pk;
      }
    }
    // Prefetch next K-step B rows into cache while WMMAs run.
    int nkt = kt + 32;
    if (nkt < K) {
      __builtin_prefetch(&Bm[(size_t)(nkt + 2 * kp) * ldb + bn + no], 0, 3);
    }
    // Wave's async copies must land before the workgroup barrier publishes
    // the LDS tile to all 8 waves.
    asm volatile("s_wait_asynccnt 0" ::: "memory");
    __syncthreads();

    // ---- A fragment (16x32 bf16, ISA 7.12.2): lane m = lane&15;
    // lanes 0-15 hold K 0..7 & 16..23, lanes 16-31 hold K 8..15 & 24..31.
    v16bf afrag[2];
#pragma unroll
    for (int tm = 0; tm < 2; ++tm) {
      int row = waveM * 32 + tm * 16 + (lane & 15);
      int kg  = (lane >> 4) * 8;
      FragCvt fc;
      fc.q[0] = *reinterpret_cast<const uint4*>(&As[row][kg]);
      fc.q[1] = *reinterpret_cast<const uint4*>(&As[row][16 + kg]);
      afrag[tm] = fc.v;
    }
    // ---- B fragment (32x16 bf16): lane n = lane&15; lanes 0-15 K=0..7 &
    // 16..23, lanes 16-31 K=8..15 & 24..31 (same K order as A). In the [n][k]
    // transposed LDS tile this is two 8-half runs -> two b128 loads.
    v16bf bfrag[4];
#pragma unroll
    for (int tn = 0; tn < 4; ++tn) {
      int col = waveN * 64 + tn * 16 + (lane & 15);
      int kh  = (lane >> 4) * 8;
      FragCvt fc;
      fc.q[0] = *reinterpret_cast<const uint4*>(&BsT[col][kh]);
      fc.q[1] = *reinterpret_cast<const uint4*>(&BsT[col][16 + kh]);
      bfrag[tn] = fc.v;
    }
#pragma unroll
    for (int tm = 0; tm < 2; ++tm)
#pragma unroll
      for (int tn = 0; tn < 4; ++tn)
        acc[tm][tn] = __builtin_amdgcn_wmma_f32_16x16x32_bf16(
            false, afrag[tm], false, bfrag[tn], (short)0, acc[tm][tn],
            false, false);
    __syncthreads();
  }

  // Epilogue. C/D layout: lanes 0-15 -> M = vgpr j, lanes 16-31 -> M = 8+j.
#pragma unroll
  for (int tm = 0; tm < 2; ++tm) {
#pragma unroll
    for (int tn = 0; tn < 4; ++tn) {
      int mBase = bm + waveM * 32 + tm * 16 + (lane >> 4) * 8;
      int nCol  = bn + waveN * 64 + tn * 16 + (lane & 15);
#pragma unroll
      for (int j = 0; j < 8; ++j) {
        int m = mBase + j;
        float v = acc[tm][tn][j];
        if (mode == 0) {
          v += bias[m];
          unsigned short bv = f2bf_bits(v);
          if (outBf)  outBf[(size_t)m * N + nCol] = bv;
          if (outBfT) outBfT[(size_t)nCol * M + m] = bv;
        } else if (mode == 1) {
          v *= rsqrtf(fmaxf(dq2[m] * dk2[nCol], EPS_F));
          outF[(size_t)m * N + nCol] = v;
        } else {
          outF[(size_t)m * N + nCol] = v;
        }
      }
    }
  }
}

// ---------------------------------------------------------------------------
// DK2[c] = sum_r Mb[r][c]^2  (column sum-of-squares, coalesced across cols)
// ---------------------------------------------------------------------------
__global__ void k_colwise_sumsq(const unsigned short* __restrict__ Mb,
                                int rows, int cols, float* __restrict__ out) {
  int c = blockIdx.x * blockDim.x + threadIdx.x;
  if (c >= cols) return;
  float s = 0.f;
  for (int r = 0; r < rows; ++r) {
    float v = bf2f(Mb[(size_t)r * cols + c]);
    s += v * v;
  }
  out[c] = s;
}

// DQ2[r] = sum_c Mb[r][c]^2  (one block per row, LDS tree reduce)
__global__ void k_rowwise_sumsq(const unsigned short* __restrict__ Mb,
                                int cols, float* __restrict__ out) {
  __shared__ float red[256];
  int r = blockIdx.x;
  float s = 0.f;
  for (int c = threadIdx.x; c < cols; c += blockDim.x) {
    float v = bf2f(Mb[(size_t)r * cols + c]);
    s += v * v;
  }
  red[threadIdx.x] = s;
  __syncthreads();
  for (int off = 128; off > 0; off >>= 1) {
    if ((int)threadIdx.x < off) red[threadIdx.x] += red[threadIdx.x + off];
    __syncthreads();
  }
  if (threadIdx.x == 0) out[r] = red[0];
}

// ---------------------------------------------------------------------------
// Column softmax stats over Y[b,k] (reduce over b, thread per k; rows are
// coalesced and the 16 MB Y lives in the 192 MB L2 for the second pass).
// ---------------------------------------------------------------------------
__global__ void k_col_softmax_stats(const float* __restrict__ Y, int Bdim,
                                    float* __restrict__ cmax,
                                    float* __restrict__ crinv) {
  int k = blockIdx.x * blockDim.x + threadIdx.x;
  if (k >= Bdim) return;
  float m = -3.4e38f;
  for (int b = 0; b < Bdim; ++b)
    m = fmaxf(m, Y[(size_t)b * Bdim + k]);
  float s = 0.f;
  for (int b = 0; b < Bdim; ++b)
    s += __expf(Y[(size_t)b * Bdim + k] - m);
  cmax[k]  = m;
  crinv[k] = 1.f / s;
}

__global__ void k_sm_to_bf16(const float* __restrict__ Y,
                             const float* __restrict__ cmax,
                             const float* __restrict__ crinv,
                             unsigned short* __restrict__ SMb, int Bdim) {
  int k = blockIdx.x * blockDim.x + threadIdx.x;
  int b = blockIdx.y;
  if (k >= Bdim) return;
  float v = __expf(Y[(size_t)b * Bdim + k] - cmax[k]) * crinv[k];
  SMb[(size_t)b * Bdim + k] = f2bf_bits(v);
}

// ---------------------------------------------------------------------------
extern "C" void kernel_launch(void* const* d_in, const int* in_sizes, int n_in,
                              void* d_out, int out_size, void* d_ws,
                              size_t ws_size, hipStream_t stream) {
  (void)n_in; (void)out_size; (void)ws_size;
  const int C = 1024, B = 2048;
  const int nBatch = in_sizes[0] / (C * B);

  const float* X    = (const float*)d_in[0];
  const float* Wk_w = (const float*)d_in[1];
  const float* Wk_b = (const float*)d_in[2];
  const float* Wq_w = (const float*)d_in[3];
  const float* Wq_b = (const float*)d_in[4];
  float* Z = (float*)d_out;

  // Workspace carve-out (~68 MB), 256B aligned slices.
  char* p = (char*)d_ws;
  auto alloc = [&](size_t bytes) {
    char* r = p;
    p += (bytes + 255) & ~(size_t)255;
    return r;
  };
  unsigned short* Wkb = (unsigned short*)alloc((size_t)C * C * 2);
  unsigned short* Wqb = (unsigned short*)alloc((size_t)C * C * 2);
  unsigned short* Xb  = (unsigned short*)alloc((size_t)nBatch * C * B * 2);
  unsigned short* Kb  = (unsigned short*)alloc((size_t)C * B * 2);   // [C,B]
  unsigned short* QT  = (unsigned short*)alloc((size_t)B * C * 2);   // [B,C]
  float* DK2   = (float*)alloc((size_t)B * 4);
  float* DQ2   = (float*)alloc((size_t)B * 4);
  float* Y     = (float*)alloc((size_t)B * B * 4);                   // [b,k]
  float* cmax  = (float*)alloc((size_t)B * 4);
  float* crinv = (float*)alloc((size_t)B * 4);
  unsigned short* SMb = (unsigned short*)alloc((size_t)B * B * 2);

  k_f32_to_bf16<<<512, 256, 0, stream>>>(Wk_w, Wkb, C * C);
  k_f32_to_bf16<<<512, 256, 0, stream>>>(Wq_w, Wqb, C * C);
  k_f32_to_bf16<<<2048, 256, 0, stream>>>(X, Xb, nBatch * C * B);

  for (int nb = 0; nb < nBatch; ++nb) {
    const unsigned short* Xn = Xb + (size_t)nb * C * B;

    // K = Wk X + bk   -> Kb [C,B] bf16
    k_wmma_gemm<<<dim3(B / 128, C / 128), 256, 0, stream>>>(
        Wkb, C, Xn, B, C, B, C, /*mode=*/0, Wk_b, Kb, nullptr, nullptr,
        nullptr, nullptr);
    // Q^T = (Wq X + bq)^T -> QT [B,C] bf16 (transposed store feeds score GEMM)
    k_wmma_gemm<<<dim3(B / 128, C / 128), 256, 0, stream>>>(
        Wqb, C, Xn, B, C, B, C, /*mode=*/0, Wq_b, nullptr, QT, nullptr,
        nullptr, nullptr);

    k_colwise_sumsq<<<B / 256, 256, 0, stream>>>(Kb, C, B, DK2);
    k_rowwise_sumsq<<<B, 256, 0, stream>>>(QT, C, DQ2);

    // Y[b,k] = (Q^T K)[b,k] * rsqrt(max(DQ2[b]*DK2[k], eps))
    k_wmma_gemm<<<dim3(B / 128, B / 128), 256, 0, stream>>>(
        QT, C, Kb, B, B, B, C, /*mode=*/1, nullptr, nullptr, nullptr, Y, DQ2,
        DK2);

    k_col_softmax_stats<<<B / 256, 256, 0, stream>>>(Y, B, cmax, crinv);
    k_sm_to_bf16<<<dim3(B / 256, B), 256, 0, stream>>>(Y, cmax, crinv, SMb, B);

    // Z = X SM  -> d_out [C,B] fp32
    k_wmma_gemm<<<dim3(B / 128, C / 128), 256, 0, stream>>>(
        Xn, B, SMb, B, C, B, B, /*mode=*/2, nullptr, nullptr, nullptr,
        Z + (size_t)nb * C * B, nullptr, nullptr);
  }
}
